// Attention_75849122447825
// MI455X (gfx1250) — matrix operand order repaired and checked by reading, not compiled, not run on hardware
//
#include <hip/hip_runtime.h>

typedef _Float16 f16;
typedef __attribute__((ext_vector_type(16))) _Float16 v16h;
typedef __attribute__((ext_vector_type(8)))  _Float16 v8h;
typedef __attribute__((ext_vector_type(8)))  float    v8f;

#define NEG_FILL_F (-1000000.0f)

// ---------- helpers ----------
__device__ inline v16h load2x8(const f16* p0, const f16* p1) {
    v8h a = *(const v8h*)p0;
    v8h b = *(const v8h*)p1;
    v16h r;
#pragma unroll
    for (int i = 0; i < 8; i++) { r[i] = a[i]; r[i + 8] = b[i]; }
    return r;
}

// ---------- 1) fp32 -> f16 conversion ----------
__global__ __launch_bounds__(256) void cvt_kernel(const float* __restrict__ in,
                                                  f16* __restrict__ out, int n) {
    int i = blockIdx.x * 256 + threadIdx.x;
    if (i < n) out[i] = (f16)in[i];
}

// ---------- 2) projections: Q = g @ wq^T, K = g @ wk^T ----------
// gh: [4096][768] f16 ; wqh/wkh: [768][768] f16 (row = h*64+z, col = d)
// Qh/Kh: [B][H][512][64] f16
__global__ __launch_bounds__(32) void proj_kernel(const f16* __restrict__ gh,
                                                  const f16* __restrict__ wqh,
                                                  const f16* __restrict__ wkh,
                                                  f16* __restrict__ Qh,
                                                  f16* __restrict__ Kh) {
    const int mt = blockIdx.x;          // 0..255 (M tiles of 16 over B*K=4096)
    const int nt = blockIdx.y;          // 0..47  (N tiles of 16 over H*Z=768)
    const int lane = threadIdx.x;       // wave32
    const int r  = lane & 15;
    const int hi = lane >> 4;

    const int abase = hi ? 16 : 0;      // A-frag half offset within 32-wide k step
    const int bbase = hi ? 16 : 0;      // B-frag k offset

    const f16* arowp = gh  + (size_t)(mt * 16 + r) * 768;
    const f16* bqp   = wqh + (size_t)(nt * 16 + r) * 768 + bbase;
    const f16* bkp   = wkh + (size_t)(nt * 16 + r) * 768 + bbase;

    v8f accq = {}; v8f acck = {};
    for (int k0 = 0; k0 < 768; k0 += 32) {
        v16h a  = load2x8(arowp + k0 + abase, arowp + k0 + abase + 8);
        v16h bq = *(const v16h*)(bqp + k0);
        v16h bk = *(const v16h*)(bkp + k0);
        accq = __builtin_amdgcn_wmma_f32_16x16x32_f16(false, a, false, bq,
                                                      (short)0, accq, false, false);
        acck = __builtin_amdgcn_wmma_f32_16x16x32_f16(false, a, false, bk,
                                                      (short)0, acck, false, false);
    }

    // D layout: lane holds col n = lane&15; VGPR v holds row m = v + 8*(lane>>4)
    const int gn = nt * 16 + r;
    const int h = gn >> 6, z = gn & 63;
#pragma unroll
    for (int v = 0; v < 8; v++) {
        int m  = v + 8 * hi;
        int gm = mt * 16 + m;
        int b = gm >> 9, kidx = gm & 511;
        size_t off = (((size_t)b * 12 + h) * 512 + kidx) * 64 + z;
        Qh[off] = (f16)accq[v];
        Kh[off] = (f16)acck[v];
    }
}

// ---------- 3) scores + sparsemax (bisection) + per-block energy ----------
__global__ __launch_bounds__(256) void attn_kernel(const f16* __restrict__ Qh,
                                                   const f16* __restrict__ Kh,
                                                   const unsigned char* __restrict__ mask,
                                                   float* __restrict__ partials) {
    __shared__ float xs[16][512];   // 32 KB score tile (x = beta * masked A)
    __shared__ float kmaskS[512];
    __shared__ float qmaskS[16];
    __shared__ float wsum[8];

    const int blk = blockIdx.x;     // 0..3071
    const int qt  = blk & 31;
    const int bh  = blk >> 5;
    const int b   = bh / 12;
    const int h   = bh % 12;

    const int tid  = threadIdx.x;
    const int wave = tid >> 5;
    const int lane = tid & 31;
    const int r  = lane & 15;
    const int hi = lane >> 4;

    for (int i = tid; i < 512; i += 256) kmaskS[i] = mask[b * 512 + i] ? 1.f : 0.f;
    if (tid < 16) qmaskS[tid] = mask[b * 512 + qt * 16 + tid] ? 1.f : 0.f;
    __syncthreads();

    const f16* qbase = Qh + (((size_t)b * 12 + h) * 512 + qt * 16) * 64;
    const f16* kbase = Kh + (((size_t)b * 12 + h) * 512) * 64;

    // A fragments for both 32-wide k steps (Z = 64)
    const int abase = hi ? 16 : 0;
    const f16* arowp = qbase + (size_t)r * 64;
    v16h a0 = load2x8(arowp + abase,      arowp + abase + 8);
    v16h a1 = load2x8(arowp + 32 + abase, arowp + 32 + abase + 8);

    const float beta = 0.125f;      // 1/sqrt(64)

    for (int t = 0; t < 4; t++) {   // each wave: 4 N-tiles -> 64 key columns
        int ntile = wave * 4 + t;
        int coln  = ntile * 16 + r;
        const f16* bp = kbase + (size_t)coln * 64 + (hi ? 16 : 0);
        v16h b0 = *(const v16h*)(bp);
        v16h b1 = *(const v16h*)(bp + 32);
        v8f acc = {};
        acc = __builtin_amdgcn_wmma_f32_16x16x32_f16(false, a0, false, b0,
                                                     (short)0, acc, false, false);
        acc = __builtin_amdgcn_wmma_f32_16x16x32_f16(false, a1, false, b1,
                                                     (short)0, acc, false, false);
        float km = kmaskS[coln];
#pragma unroll
        for (int v = 0; v < 8; v++) {
            int m = v + 8 * hi;
            float on = km * qmaskS[m];
            xs[m][coln] = beta * ((on != 0.f) ? acc[v] : NEG_FILL_F);
        }
    }
    __syncthreads();

    // sparsemax per row via bisection on tau: sum(max(x - tau, 0)) == 1
    float esum = 0.f;
    for (int rr = 0; rr < 2; rr++) {
        int row = wave * 2 + rr;
        float x[16];
#pragma unroll
        for (int i = 0; i < 16; i++) x[i] = xs[row][lane * 16 + i];

        float mx = x[0];
#pragma unroll
        for (int i = 1; i < 16; i++) mx = fmaxf(mx, x[i]);
        for (int s = 16; s >= 1; s >>= 1) mx = fmaxf(mx, __shfl_xor(mx, s, 32));

        float lo = mx - 1.f, hib = mx;       // s(lo) >= 1, s(hib) == 0
        for (int it = 0; it < 32; it++) {
            float mid = 0.5f * (lo + hib);
            float s = 0.f;
#pragma unroll
            for (int i = 0; i < 16; i++) s += fmaxf(x[i] - mid, 0.f);
            for (int sh = 16; sh >= 1; sh >>= 1) s += __shfl_xor(s, sh, 32);
            if (s >= 1.f) lo = mid; else hib = mid;
        }
        float tau = 0.5f * (lo + hib);

        float sd = 0.f, sp = 0.f;
#pragma unroll
        for (int i = 0; i < 16; i++) {
            float p = fmaxf(x[i] - tau, 0.f);
            sd += x[i] * p;
            sp += p * p;
        }
        for (int sh = 16; sh >= 1; sh >>= 1) {
            sd += __shfl_xor(sd, sh, 32);
            sp += __shfl_xor(sp, sh, 32);
        }
        esum += sd - sqrtf(sp);
    }
    if (lane == 0) wsum[wave] = esum;
    __syncthreads();
    if (tid == 0) {
        float t = 0.f;
        for (int w = 0; w < 8; w++) t += wsum[w];
        partials[blk] = t;
    }
}

// ---------- 4) final reduction (fp64 accumulation) ----------
__global__ __launch_bounds__(256) void reduce_kernel(const float* __restrict__ partials,
                                                     int n, float* __restrict__ out) {
    __shared__ double sh[256];
    double s = 0.0;
    for (int i = threadIdx.x; i < n; i += 256) s += (double)partials[i];
    sh[threadIdx.x] = s;
    __syncthreads();
    for (int stride = 128; stride > 0; stride >>= 1) {
        if (threadIdx.x < stride) sh[threadIdx.x] += sh[threadIdx.x + stride];
        __syncthreads();
    }
    if (threadIdx.x == 0) out[0] = (float)(-8.0 * sh[0]);   // e / beta, beta = 1/8
}

extern "C" void kernel_launch(void* const* d_in, const int* in_sizes, int n_in,
                              void* d_out, int out_size, void* d_ws, size_t ws_size,
                              hipStream_t stream) {
    const float* g  = (const float*)d_in[0];                 // [8,512,768]
    const float* wq = (const float*)d_in[1];                 // [12,64,768]
    const float* wk = (const float*)d_in[2];                 // [12,64,768]
    const unsigned char* mask = (const unsigned char*)d_in[3]; // [8,512] bool

    const int NG = 8 * 512 * 768;      // 3,145,728
    const int NW = 12 * 64 * 768;      //   589,824
    const int NQK = 8 * 12 * 512 * 64; // 3,145,728
    const int NBLK = 8 * 12 * 32;      // 3,072

    char* ws = (char*)d_ws;
    size_t off = 0;
    auto alloc = [&](size_t bytes) { char* p = ws + off; off = (off + bytes + 255) & ~(size_t)255; return p; };
    f16*   gh  = (f16*)alloc((size_t)NG * 2);
    f16*   wqh = (f16*)alloc((size_t)NW * 2);
    f16*   wkh = (f16*)alloc((size_t)NW * 2);
    f16*   Qh  = (f16*)alloc((size_t)NQK * 2);
    f16*   Kh  = (f16*)alloc((size_t)NQK * 2);
    float* partials = (float*)alloc((size_t)NBLK * 4);
    (void)ws_size; (void)in_sizes; (void)n_in; (void)out_size;

    cvt_kernel<<<(NG + 255) / 256, 256, 0, stream>>>(g,  gh,  NG);
    cvt_kernel<<<(NW + 255) / 256, 256, 0, stream>>>(wq, wqh, NW);
    cvt_kernel<<<(NW + 255) / 256, 256, 0, stream>>>(wk, wkh, NW);

    proj_kernel<<<dim3(256, 48), 32, 0, stream>>>(gh, wqh, wkh, Qh, Kh);
    attn_kernel<<<NBLK, 256, 0, stream>>>(Qh, Kh, mask, partials);
    reduce_kernel<<<1, 256, 0, stream>>>(partials, NBLK, (float*)d_out);
}
